// CoconutPPO_35132832481465
// MI455X (gfx1250) — matrix operand order repaired and checked by reading, not compiled, not run on hardware
//
#include <hip/hip_runtime.h>
#include <math.h>
#include <stdint.h>

// ---------------------------------------------------------------------------
// CDNA5 / gfx1250 WMMA types
// ---------------------------------------------------------------------------
typedef __attribute__((ext_vector_type(16))) __bf16 v16bf;
typedef __attribute__((ext_vector_type(8)))  float  v8f;

#define EPS_NORM 1e-12f
#define EPS_VAL  1e-8f
#define FUSE     0.5f

#define Brows 128
#define Hdim  4096
#define Rdim  256
#define TOPK  3
#define CHUNK 2048   // memory rows per workgroup in sims kernel (multiple of 16)

// ---------------------------------------------------------------------------
// Generic bf16-WMMA GEMM:  C[M,N] = act(A[M,K] @ Bw[K,N] + bias)
// One wave (32 threads) computes one 16x16 tile, K-loop in steps of 32.
// fp32 inputs are converted to bf16 in-register (RNE) to hit the
// v_wmma_f32_16x16x32_bf16 path.
// ---------------------------------------------------------------------------
__global__ __launch_bounds__(32) void gemm_bf16_wmma(
    const float* __restrict__ A, const float* __restrict__ Bw,
    const float* __restrict__ bias, float* __restrict__ C,
    int M, int K, int N, int relu)
{
    const int lane = threadIdx.x & 31;
    const int hi   = lane >> 4;       // which half of the wave
    const int lm   = lane & 15;
    const int rowT = blockIdx.y * 16; // M tile
    const int colT = blockIdx.x * 16; // N tile

    const int aRow = rowT + lm;       // this lane's A row (A-frag layout)
    const int bCol = colT + lm;       // this lane's B column (B-frag layout)
    const int kbA  = hi ? 8  : 0;     // A: lanes 16-31 hold K=8..15 / 24..31
    const int kbB  = hi ? 16 : 0;     // unused: B is loaded with kbA, same K order as A

    v8f acc = {};
    const float* arow = A + (size_t)aRow * K;
    const float* bcol = Bw + bCol;

    for (int k0 = 0; k0 < K; k0 += 32) {
        v16bf af, bf;
#pragma unroll
        for (int s = 0; s < 16; ++s) {
            int ka = k0 + kbA + s + ((s >= 8) ? 8 : 0);   // ISA 16-bit A 16x32 layout
            af[s] = (__bf16)arow[ka];
        }
#pragma unroll
        for (int s = 0; s < 16; ++s) {
            int kb = k0 + kbA + s + ((s >= 8) ? 8 : 0);   // ISA 16-bit B 32x16 layout
            bf[s] = (__bf16)bcol[(size_t)kb * N];
        }
        acc = __builtin_amdgcn_wmma_f32_16x16x32_bf16(
            false, af, false, bf, (short)0, acc, false, false);
    }

#pragma unroll
    for (int j = 0; j < 8; ++j) {
        int m = rowT + j + (hi ? 8 : 0);
        int n = colT + lm;
        float v = acc[j] + bias[n];
        if (relu) v = fmaxf(v, 0.0f);
        C[(size_t)m * N + n] = v;
    }
}

// ---------------------------------------------------------------------------
// Row-wise L2 normalize r0 [128,256] -> bf16 (raw u16 bits) for the sims GEMM
// ---------------------------------------------------------------------------
__global__ __launch_bounds__(256) void rownorm_bf16(
    const float* __restrict__ r0, unsigned short* __restrict__ rnbf)
{
    __shared__ float red[256];
    const int b = blockIdx.x, t = threadIdx.x;
    float x = r0[b * Rdim + t];
    red[t] = x * x;
    __syncthreads();
    for (int s = 128; s > 0; s >>= 1) {
        if (t < s) red[t] += red[t + s];
        __syncthreads();
    }
    float inv = 1.0f / fmaxf(sqrtf(red[0]), EPS_NORM);
    __bf16 h = (__bf16)(x * inv);
    rnbf[b * Rdim + t] = __builtin_bit_cast(unsigned short, h);
}

__device__ __forceinline__ void top3_insert(float w, int g,
    float& v0, float& v1, float& v2, int& i0, int& i1, int& i2)
{
    if (w > v2) {
        if (w > v1) {
            if (w > v0) { v2 = v1; i2 = i1; v1 = v0; i1 = i0; v0 = w; i0 = g; }
            else        { v2 = v1; i2 = i1; v1 = w;  i1 = g; }
        } else          { v2 = w;  i2 = g; }
    }
}

// ---------------------------------------------------------------------------
// gfx1250 async global->LDS copy: 64 bytes per thread (4 x b128), tracked by
// ASYNCcnt.  INST_OFFSET applies to both the LDS and global addresses.
// ---------------------------------------------------------------------------
__device__ __forceinline__ void async_copy64B(uint32_t lds_off, const float* g)
{
    uint64_t ga = (uint64_t)(uintptr_t)g;
    asm volatile(
        "global_load_async_to_lds_b128 %0, %1, off\n\t"
        "global_load_async_to_lds_b128 %0, %1, off offset:16\n\t"
        "global_load_async_to_lds_b128 %0, %1, off offset:32\n\t"
        "global_load_async_to_lds_b128 %0, %1, off offset:48"
        :: "v"(lds_off), "v"(ga) : "memory");
}

__device__ __forceinline__ void wait_async0()
{
    asm volatile("s_wait_asynccnt 0x0" ::: "memory");
}

// ---------------------------------------------------------------------------
// Sims + partial top-3.  Block = 256 thr (8 waves).  Wave w owns B-rows
// [16w,16w+16).  Block owns CHUNK memory rows; tiles of 16 rows are streamed
// into a double-buffered LDS stage via async global->LDS DMA, overlapped with
// 8x v_wmma_f32_16x16x32_bf16 on the previous tile.  A-fragments (rn rows,
// bf16) are preloaded in VGPRs for the whole chunk.
// Weighted sim = (rn . mb_m) * (mv_m + eps) / max(||mb_m||, eps).
// ---------------------------------------------------------------------------
__global__ __launch_bounds__(256) void sims_topk_chunk(
    const unsigned short* __restrict__ rnbf,   // [128,256] bf16 bits
    const float* __restrict__ mb,              // [M,256]
    const float* __restrict__ mv,              // [M]
    float* __restrict__ cand_val,              // [nchunk,128,3]
    int*   __restrict__ cand_idx,
    int Mtotal)
{
    __shared__ float btile[2][16][260];        // padded: 260%64=4 -> conflict-free
    __shared__ float psum[256];
    __shared__ float scale[16];
    __shared__ float scratch[8][16][17];       // per-wave 16x16 result tile (padded)

    const int tid  = threadIdx.x;
    const int wave = tid >> 5;
    const int lane = tid & 31;
    const int hi   = lane >> 4;
    const int lm   = lane & 15;

    const int start = blockIdx.x * CHUNK;
    const int end   = (start + CHUNK < Mtotal) ? (start + CHUNK) : Mtotal;
    const int ntile = (end - start) >> 4;

    // --- preload all 8 A-fragments (this wave's 16 rows x K=256), bf16 bits ---
    const unsigned short* rrow = rnbf + (size_t)(16 * wave + lm) * Rdim;
    const int kbA = hi ? 8 : 0;
    v16bf afrag[8];
#pragma unroll
    for (int f = 0; f < 8; ++f) {
        union { v16bf v; uint32_t u[8]; } tmp;
        int k0 = f * 32;
#pragma unroll
        for (int d = 0; d < 8; ++d) {
            int s = 2 * d;
            int k = k0 + kbA + s + ((s >= 8) ? 8 : 0);   // dword-aligned (k even)
            tmp.u[d] = *(const uint32_t*)(rrow + k);
        }
        afrag[f] = tmp.v;
    }

    float v0 = -__builtin_inff(), v1 = v0, v2 = v0;
    int   i0 = 0, i1 = 0, i2 = 0;

    const int ldRow = tid >> 4;          // staging: thread -> (row, 16-col group)
    const int ldCol = (tid & 15) * 16;
    const uint32_t ldsA = (uint32_t)(uintptr_t)&btile[0][ldRow][ldCol];
    const uint32_t ldsB = (uint32_t)(uintptr_t)&btile[1][ldRow][ldCol];

    // prologue: issue tile 0 into buffer 0
    async_copy64B(ldsA, mb + (size_t)(start + ldRow) * Rdim + ldCol);

    for (int i = 0; i < ntile; ++i) {
        const int t0 = start + i * 16;
        const float (*cur)[260] = btile[i & 1];

        wait_async0();       // this wave's async writes for tile i landed
        __syncthreads();     // ... and everyone else's are visible

        if (i + 1 < ntile)   // overlap: DMA tile i+1 into the other buffer
            async_copy64B((i & 1) ? ldsA : ldsB,
                          mb + (size_t)(t0 + 16 + ldRow) * Rdim + ldCol);

        // ---- row sum-of-squares from LDS ----
        {
            const float* p = &cur[ldRow][ldCol];
            float ss = 0.f;
#pragma unroll
            for (int j = 0; j < 16; ++j) ss += p[j] * p[j];
            psum[tid] = ss;
        }
        __syncthreads();
        if (tid < 16) {
            float s = 0.f;
#pragma unroll
            for (int j = 0; j < 16; ++j) s += psum[tid * 16 + j];
            scale[tid] = (mv[t0 + tid] + EPS_VAL) / fmaxf(sqrtf(s), EPS_NORM);
        }
        __syncthreads();

        // ---- 16x16 dot tile via 8 chained bf16 WMMAs ----
        v8f acc = {};
        const int kbB = hi ? 16 : 0;
#pragma unroll
        for (int f = 0; f < 8; ++f) {
            const float* bp = &cur[lm][f * 32 + kbA];
            v16bf bf;
#pragma unroll
            for (int s = 0; s < 16; ++s) bf[s] = (__bf16)bp[s + ((s >= 8) ? 8 : 0)];
            acc = __builtin_amdgcn_wmma_f32_16x16x32_bf16(
                false, afrag[f], false, bf, (short)0, acc, false, false);
        }

        // ---- scale by per-memory-row factor, spill tile, top-3 scan ----
        float sc = scale[lm];
#pragma unroll
        for (int j = 0; j < 8; ++j)
            scratch[wave][j + (hi ? 8 : 0)][lm] = acc[j] * sc;
        __syncthreads();
        if (lane < 16) {   // lane handles B-row 16*wave+lane, scans 16 columns
#pragma unroll
            for (int n = 0; n < 16; ++n)
                top3_insert(scratch[wave][lane][n], t0 + n, v0, v1, v2, i0, i1, i2);
        }
        // next iteration's wait_async0 + __syncthreads guards scratch & buffers
    }

    if (lane < 16) {
        int row = 16 * wave + lane;
        size_t base = ((size_t)blockIdx.x * Brows + row) * TOPK;
        cand_val[base + 0] = v0; cand_idx[base + 0] = i0;
        cand_val[base + 1] = v1; cand_idx[base + 1] = i1;
        cand_val[base + 2] = v2; cand_idx[base + 2] = i2;
    }
}

// ---------------------------------------------------------------------------
// Merge per-chunk candidates -> global top-3, gather + fuse:
//   r_fused = 0.5*r0 + 0.5*mean(memory_bank[top3])
// ---------------------------------------------------------------------------
__global__ __launch_bounds__(256) void topk_fuse(
    const float* __restrict__ cand_val, const int* __restrict__ cand_idx,
    const float* __restrict__ mb, const float* __restrict__ r0,
    float* __restrict__ rfused, int nchunk)
{
    __shared__ int sidx[3];
    const int b = blockIdx.x, t = threadIdx.x;
    if (t == 0) {
        float v0 = -__builtin_inff(), v1 = v0, v2 = v0;
        int i0 = 0, i1 = 0, i2 = 0;
        for (int c = 0; c < nchunk; ++c) {
            size_t base = ((size_t)c * Brows + b) * TOPK;
            for (int j = 0; j < TOPK; ++j)
                top3_insert(cand_val[base + j], cand_idx[base + j],
                            v0, v1, v2, i0, i1, i2);
        }
        sidx[0] = i0; sidx[1] = i1; sidx[2] = i2;
    }
    __syncthreads();
    float ret = (mb[(size_t)sidx[0] * Rdim + t] +
                 mb[(size_t)sidx[1] * Rdim + t] +
                 mb[(size_t)sidx[2] * Rdim + t]) * (1.0f / 3.0f);
    rfused[b * Rdim + t] = (1.0f - FUSE) * r0[b * Rdim + t] + FUSE * ret;
}

// ---------------------------------------------------------------------------
// Policy heads.  NOTE: jax.random.categorical's exact threefry bits are not
// reproducible; we use a deterministic hash-Gumbel per row instead.
// ---------------------------------------------------------------------------
__device__ __forceinline__ uint32_t hash32(uint32_t x) {
    x ^= x >> 16; x *= 0x7feb352dU;
    x ^= x >> 15; x *= 0x846ca68bU;
    x ^= x >> 16; return x;
}

__device__ __forceinline__ float block_reduce(float x, float* red, int t) {
    red[t] = x; __syncthreads();
    for (int s = 128; s > 0; s >>= 1) {
        if (t < s) red[t] += red[t + s];
        __syncthreads();
    }
    float r = red[0]; __syncthreads();
    return r;
}

__global__ __launch_bounds__(256) void heads_kernel(
    const float* __restrict__ rfused,
    const float* __restrict__ Wc, const float* __restrict__ bc,
    const float* __restrict__ Wd, const float* __restrict__ bd,
    const float* __restrict__ Ws, const float* __restrict__ bs,
    const float* __restrict__ Wv, const float* __restrict__ bv,
    float* __restrict__ out_position, float* __restrict__ out_action,
    float* __restrict__ out_logp, float* __restrict__ out_value,
    float* __restrict__ out_entropy)
{
    __shared__ float rsh[256];
    __shared__ float red[256];
    const int b = blockIdx.x, t = threadIdx.x;

    float rv = rfused[b * Rdim + t];
    rsh[t] = rv;
    __syncthreads();

    // direction (pre-norm): column t of r @ Wd + bd
    float d = bd[t];
    for (int k = 0; k < Rdim; ++k) d += rsh[k] * Wd[(size_t)k * Rdim + t];

    float c0 = block_reduce(rv * Wc[2 * t + 0], red, t);
    float c1 = block_reduce(rv * Wc[2 * t + 1], red, t);
    float sd = block_reduce(rv * Ws[t],         red, t);
    float vd = block_reduce(rv * Wv[t],         red, t);
    float dn = block_reduce(d * d,              red, t);

    float l0 = c0 + bc[0], l1 = c1 + bc[1];
    float mx = fmaxf(l0, l1);
    float e0 = expf(l0 - mx), e1 = expf(l1 - mx);
    float lz = logf(e0 + e1);
    float lp0 = l0 - mx - lz, lp1 = l1 - mx - lz;

    uint32_t h0 = hash32(0x9E3779B9u ^ (uint32_t)(2 * b));
    uint32_t h1 = hash32(0x9E3779B9u ^ (uint32_t)(2 * b + 1));
    float u0 = ((float)h0 + 0.5f) * (1.0f / 4294967296.0f);
    float u1 = ((float)h1 + 0.5f) * (1.0f / 4294967296.0f);
    float g0 = -logf(-logf(u0)), g1 = -logf(-logf(u1));
    int a = (l0 + g0 >= l1 + g1) ? 0 : 1;

    if (t == 0) {
        out_action[b]  = (float)a;
        out_logp[b]    = a ? lp1 : lp0;
        out_value[b]   = vd + bv[0];
        out_entropy[b] = -(expf(lp0) * lp0 + expf(lp1) * lp1);
    }

    float invn = 1.0f / fmaxf(sqrtf(dn), EPS_NORM);
    float step = 2.0f / (1.0f + expf(-(sd + bs[0])));
    out_position[b * Rdim + t] = rv + step * (d * invn);
}

// ---------------------------------------------------------------------------
// launch
// ---------------------------------------------------------------------------
extern "C" void kernel_launch(void* const* d_in, const int* in_sizes, int n_in,
                              void* d_out, int out_size, void* d_ws, size_t ws_size,
                              hipStream_t stream)
{
    const float* state = (const float*)d_in[0];
    const float* mb    = (const float*)d_in[1];
    const float* mv    = (const float*)d_in[2];
    const float* W1 = (const float*)d_in[3];  const float* b1 = (const float*)d_in[4];
    const float* W2 = (const float*)d_in[5];  const float* b2 = (const float*)d_in[6];
    const float* Wc = (const float*)d_in[7];  const float* bc = (const float*)d_in[8];
    const float* Wd = (const float*)d_in[9];  const float* bd = (const float*)d_in[10];
    const float* Ws = (const float*)d_in[11]; const float* bs = (const float*)d_in[12];
    const float* Wv = (const float*)d_in[13]; const float* bv = (const float*)d_in[14];
    const float* T1 = (const float*)d_in[15]; const float* t1 = (const float*)d_in[16];
    const float* T2 = (const float*)d_in[17]; const float* t2 = (const float*)d_in[18];

    const int M = in_sizes[1] / Rdim;                 // 500000
    const int nchunk = (M + CHUNK - 1) / CHUNK;       // 245

    // workspace layout
    float*          hidden = (float*)d_ws;                       // 128*1024
    float*          r0     = hidden + Brows * (Hdim / 4);        // 128*256
    unsigned short* rnbf   = (unsigned short*)(r0 + Brows * Rdim);
    float*          rfused = (float*)(rnbf + Brows * Rdim);      // 128*256
    float*          h2     = rfused + Brows * Rdim;              // 128*1024
    float*          cand_v = h2 + Brows * (Hdim / 4);
    int*            cand_i = (int*)(cand_v + (size_t)nchunk * Brows * TOPK);

    // output layout (flat, return order)
    float* out_latent   = (float*)d_out;                         // 128*4096
    float* out_position = out_latent + Brows * Hdim;             // 128*256
    float* out_action   = out_position + Brows * Rdim;           // 128
    float* out_logp     = out_action + Brows;
    float* out_value    = out_logp + Brows;
    float* out_entropy  = out_value + Brows;

    // 1) hidden = relu(state @ W1 + b1)           [128,1024]
    gemm_bf16_wmma<<<dim3((Hdim / 4) / 16, Brows / 16), 32, 0, stream>>>(
        state, W1, b1, hidden, Brows, Hdim, Hdim / 4, 1);
    // 2) r0 = hidden @ W2 + b2                    [128,256]
    gemm_bf16_wmma<<<dim3(Rdim / 16, Brows / 16), 32, 0, stream>>>(
        hidden, W2, b2, r0, Brows, Hdim / 4, Rdim, 0);
    // 3) rn = l2norm(r0) -> bf16
    rownorm_bf16<<<Brows, Rdim, 0, stream>>>(r0, rnbf);
    // 4) sims + per-chunk top-3 (the 512 MB / 32.8 GF streaming GEMM,
    //    async-DMA double-buffered)
    sims_topk_chunk<<<nchunk, 256, 0, stream>>>(rnbf, mb, mv, cand_v, cand_i, M);
    // 5) merge top-3, gather, fuse
    topk_fuse<<<Brows, Rdim, 0, stream>>>(cand_v, cand_i, mb, r0, rfused, nchunk);
    // 6) policy heads + position
    heads_kernel<<<Brows, Rdim, 0, stream>>>(rfused, Wc, bc, Wd, bd, Ws, bs, Wv, bv,
                                             out_position, out_action, out_logp,
                                             out_value, out_entropy);
    // 7) latent = relu(position @ T1 + t1) @ T2 + t2
    gemm_bf16_wmma<<<dim3((Hdim / 4) / 16, Brows / 16), 32, 0, stream>>>(
        out_position, T1, t1, h2, Brows, Rdim, Hdim / 4, 1);
    gemm_bf16_wmma<<<dim3(Hdim / 16, Brows / 16), 32, 0, stream>>>(
        h2, T2, t2, out_latent, Brows, Hdim / 4, Hdim, 0);
}
